// SelfAttentionDiff_47072841564515
// MI455X (gfx1250) — matrix operand order repaired and checked by reading, not compiled, not run on hardware
//
#include <hip/hip_runtime.h>
#include <hip/hip_bf16.h>

typedef __attribute__((ext_vector_type(16))) _Float16 v16h;
typedef __attribute__((ext_vector_type(8)))  _Float16 v8h;
typedef __attribute__((ext_vector_type(8)))  float    v8f;
typedef __attribute__((ext_vector_type(4)))  unsigned int u32x4;
typedef __attribute__((ext_vector_type(8)))  int          i32x8;
typedef __attribute__((ext_vector_type(4)))  int          i32x4;

#define BATCH 8
#define CH    256
#define HW    4096
#define PLANE (CH * HW)          // 1048576 elements per batch plane

// ---------------- workspace layout (bytes) ----------------
#define OFF_W16   ((size_t)0)          // 4 * 256*256 f16  (524288 B)
#define OFF_STATS ((size_t)524288)     // 8*32*{mean,rstd} f32
#define OFF_XNT   ((size_t)1048576)    // (b, n, c) f16  16 MB
#define OFF_Q     ((size_t)17825792)   // (b, n, c) f16  (scaled by 1/sqrt c)
#define OFF_K     ((size_t)34603008)   // (b, n, c) f16
#define OFF_V     ((size_t)51380224)   // (b, c, n) f16
#define OFF_H     ((size_t)68157440)   // (b, n, c) f16 attention output

__device__ __forceinline__ v8f wmma_f16(v16h a, v16h b, v8f c) {
    return __builtin_amdgcn_wmma_f32_16x16x32_f16(
        /*neg_a=*/false, a, /*neg_b=*/false, b,
        /*c_mod=*/(short)0, c, /*reuse_a=*/false, /*reuse_b=*/false);
}

// A fragment (16x32 f16, M rows x K cols) from row-major [row][K] source.
// lane = M row; lanes 0-15 hold K {0..7,16..23}, lanes 16-31 hold K {8..15,24..31}
__device__ __forceinline__ v16h load_a16(const _Float16* base, int row_stride,
                                         int r0, int k0) {
    int l = threadIdx.x & 31;
    const _Float16* row = base + (size_t)(r0 + (l & 15)) * row_stride + k0 + ((l >> 4) * 8);
    union { v16h v; v8h h8[2]; } u;
    u.h8[0] = *(const v8h*)(row);
    u.h8[1] = *(const v8h*)(row + 16);
    return u.v;
}

// B fragment (32x16 f16, K rows x N cols) from row-major [N][K] source
// (K contiguous per column). lane%16 = N col; lanes 0-15 K {0..7,16..23}, 16-31 K {8..15,24..31}.
__device__ __forceinline__ v16h load_b16(const _Float16* base, int row_stride,
                                         int n0, int k0) {
    int l = threadIdx.x & 31;
    const _Float16* row = base + (size_t)(n0 + (l & 15)) * row_stride + k0 + ((l >> 4) * 8);
    union { v16h v; v8h h8[2]; } u;
    u.h8[0] = *(const v8h*)(row);
    u.h8[1] = *(const v8h*)(row + 16);
    return u.v;
}

// ---------------- Tensor Data Mover: 2-D tile Global -> LDS ----------------
// D# per CDNA5 ISA 8.3/8.4. Elements are f16 (data_size code 1 = 2 bytes).
// tile_d0 elements contiguous along dim0, tile_d1 rows, row stride stride0 (elements).
__device__ __forceinline__ void tdm_load_2d(const _Float16* gaddr, unsigned lds_off,
                                            unsigned tile_d0, unsigned tile_d1,
                                            unsigned tensor_d0, unsigned tensor_d1,
                                            unsigned long long stride0) {
    unsigned long long ga = (unsigned long long)(size_t)gaddr;
    u32x4 g0;
    g0[0] = 1u;                                              // count=1, user D#
    g0[1] = lds_off;                                         // lds_addr (bytes)
    g0[2] = (unsigned)(ga & 0xffffffffu);                    // global_addr[31:0]
    g0[3] = (unsigned)((ga >> 32) & 0x1ffffffu) | (2u << 30); // ga[56:32], type=2
    i32x8 g1;
    g1[0] = (int)(1u << 16);                                 // data_size=1 (2B)
    g1[1] = (int)((tensor_d0 & 0xffffu) << 16);              // tensor_dim0[15:0]
    g1[2] = (int)(((tensor_d0 >> 16) & 0xffffu) | ((tensor_d1 & 0xffffu) << 16));
    g1[3] = (int)(((tensor_d1 >> 16) & 0xffffu) | ((tile_d0 & 0xffffu) << 16));
    g1[4] = (int)(tile_d1 & 0xffffu);                        // tile_dim1; tile_dim2=0
    g1[5] = (int)(unsigned)(stride0 & 0xffffffffu);          // dim0 stride [31:0]
    g1[6] = (int)((stride0 >> 32) & 0xffffu);                // dim0 stride [47:32]
    g1[7] = 0;                                               // dim1 stride unused (2D)
    i32x4 z4 = {0, 0, 0, 0};
#if __clang_major__ >= 23
    i32x8 z8 = {0, 0, 0, 0, 0, 0, 0, 0};
    __builtin_amdgcn_tensor_load_to_lds(g0, g1, z4, z4, z8, 0);
#else
    __builtin_amdgcn_tensor_load_to_lds(g0, g1, z4, z4, 0);
#endif
}

// ---------------- weight conversion (fold 1/sqrt(c) into wq) ----------------
__global__ void conv_w_kernel(const float* wq, const float* wk, const float* wv,
                              const float* wp, _Float16* w16) {
    int idx = blockIdx.x * 256 + threadIdx.x;          // 65536 per matrix
    w16[idx]              = (_Float16)(wq[idx] * 0.0625f);
    w16[idx + 65536]      = (_Float16)wk[idx];
    w16[idx + 131072]     = (_Float16)wv[idx];
    w16[idx + 196608]     = (_Float16)wp[idx];
}

// ---------------- group norm stats: one block per (b, g) ----------------
__global__ void gn_stats_kernel(const float* x, float* stats) {
    int bg = blockIdx.x;                 // b*32 + g
    int b = bg >> 5, g = bg & 31;
    const float* base = x + ((size_t)b * CH + g * 8) * HW;
    float s = 0.f, ss = 0.f;
    for (int i = threadIdx.x; i < 8 * HW; i += 256) {
        float v = base[i];
        s += v; ss += v * v;
    }
    __shared__ float sh0[256], sh1[256];
    sh0[threadIdx.x] = s; sh1[threadIdx.x] = ss;
    __syncthreads();
    for (int off = 128; off > 0; off >>= 1) {
        if ((int)threadIdx.x < off) {
            sh0[threadIdx.x] += sh0[threadIdx.x + off];
            sh1[threadIdx.x] += sh1[threadIdx.x + off];
        }
        __syncthreads();
    }
    if (threadIdx.x == 0) {
        float mean = sh0[0] * (1.f / 32768.f);
        float var  = sh1[0] * (1.f / 32768.f) - mean * mean;
        stats[bg * 2 + 0] = mean;
        stats[bg * 2 + 1] = rsqrtf(var + 1e-5f);
    }
}

// ---------------- apply GN, write xn transposed (b, n, c) f16 ----------------
__global__ void gn_apply_kernel(const float* x, const float* gnw, const float* gnb,
                                const float* stats, _Float16* xnt) {
    size_t idx = (size_t)blockIdx.x * 256 + threadIdx.x;  // over 8*256*4096
    int n = (int)(idx & (HW - 1));
    size_t t = idx >> 12;
    int c = (int)(t & (CH - 1));
    int b = (int)(t >> 8);
    int g = c >> 3;
    float mean = stats[(b * 32 + g) * 2 + 0];
    float rstd = stats[(b * 32 + g) * 2 + 1];
    float v = (x[idx] - mean) * rstd * gnw[c] + gnb[c];
    xnt[((size_t)b * HW + n) * CH + c] = (_Float16)v;
}

// ---------------- QKV projection GEMM (WMMA) ----------------
// grid (M/32=8, N/256=16, b*3=24), block 256 (8 waves); wave computes 16x64.
__global__ void __launch_bounds__(256) qkv_gemm_kernel(
        const _Float16* w16, const _Float16* xnt,
        const float* bq, const float* bk, const float* bv,
        _Float16* q16, _Float16* k16, _Float16* v16) {
    int wave = threadIdx.x >> 5, l = threadIdx.x & 31;
    int which = blockIdx.z % 3;
    int b     = blockIdx.z / 3;
    int o0 = blockIdx.x * 32 + (wave & 1) * 16;
    int n0 = blockIdx.y * 256 + (wave >> 1) * 64;
    const _Float16* A  = w16 + which * 65536;
    const float*    bi = (which == 0) ? bq : (which == 1) ? bk : bv;
    const _Float16* Bm = xnt + (size_t)b * PLANE;

    v8f acc[4] = {};
    for (int k = 0; k < CH; k += 32) {
        v16h a = load_a16(A, CH, o0, k);
#pragma unroll
        for (int t = 0; t < 4; t++)
            acc[t] = wmma_f16(a, load_b16(Bm, CH, n0 + t * 16, k), acc[t]);
    }
    float bscale = (which == 0) ? 0.0625f : 1.f;
    int colh = l >> 4, col = l & 15;
#pragma unroll
    for (int t = 0; t < 4; t++) {
#pragma unroll
        for (int r = 0; r < 8; r++) {
            int m = o0 + r + 8 * colh;
            int n = n0 + t * 16 + col;
            float vv = acc[t][r] + bi[m] * bscale;
            if (which == 2) {
                v16[(size_t)b * PLANE + (size_t)m * HW + n] = (_Float16)vv;  // (c, n)
            } else {
                _Float16* dst = which ? k16 : q16;
                dst[(size_t)b * PLANE + (size_t)n * CH + m] = (_Float16)vv;  // (n, c)
            }
        }
    }
}

// ---------------- flash attention ----------------
// grid (HW/64=64, b=8), block 128 (4 waves); wave owns a 16-row tile.
// K/V tiles staged once per block via TDM (double-buffered), fragments from LDS.
__global__ void __launch_bounds__(128) flash_attn_kernel(
        const _Float16* q16, const _Float16* k16, const _Float16* v16,
        _Float16* ht) {
    int wv_id = threadIdx.x >> 5, l = threadIdx.x & 31;
    int b  = blockIdx.y;
    int i0 = blockIdx.x * 64 + wv_id * 16;
    const _Float16* qb = q16 + (size_t)b * PLANE;
    const _Float16* kb = k16 + (size_t)b * PLANE;
    const _Float16* vb = v16 + (size_t)b * PLANE;

    __shared__ _Float16 kbuf[2][32 * 256];  // 32 j-rows x 256 c   (16 KB each)
    __shared__ _Float16 vbuf[2][256 * 32];  // 256 c-rows x 32 j   (16 KB each)
    __shared__ _Float16 pbuf[4][16 * 32];   // per-wave P transpose buffer
    _Float16* pb = pbuf[wv_id];

    // Q A-fragments for all of K=256, kept resident (64 VGPRs)
    v16h aq[8];
#pragma unroll
    for (int k = 0; k < 8; k++) aq[k] = load_a16(qb, CH, i0, k * 32);

    v8f o[16] = {};                 // 16 rows x 256 cols fp32 accumulator
    float mrow[8], lrow[8];
#pragma unroll
    for (int r = 0; r < 8; r++) { mrow[r] = -1e30f; lrow[r] = 0.f; }

    // prologue: stage tile 0
    if (wv_id == 0) {
        tdm_load_2d(kb, (unsigned)(size_t)&kbuf[0][0], CH, 32, CH, HW, CH);
        tdm_load_2d(vb, (unsigned)(size_t)&vbuf[0][0], 32, CH, HW, CH, HW);
        __builtin_amdgcn_s_wait_tensorcnt(0);
    }
    __syncthreads();

    int colh = l >> 4, col = l & 15;
    for (int jt = 0; jt < HW; jt += 32) {
        int cur = (jt >> 5) & 1;
        // kick off next tile while computing on the current one
        if (wv_id == 0 && jt + 32 < HW) {
            tdm_load_2d(kb + (size_t)(jt + 32) * CH,
                        (unsigned)(size_t)&kbuf[1 - cur][0], CH, 32, CH, HW, CH);
            tdm_load_2d(vb + (jt + 32),
                        (unsigned)(size_t)&vbuf[1 - cur][0], 32, CH, HW, CH, HW);
        }
        const _Float16* kt = kbuf[cur];
        const _Float16* vt = vbuf[cur];

        // S tiles: rows i0..i0+15, cols jt..jt+31 (fp32), K fragments from LDS
        v8f s0 = {}, s1 = {};
#pragma unroll
        for (int k = 0; k < 8; k++) {
            s0 = wmma_f16(aq[k], load_b16(kt, CH, 0,  k * 32), s0);
            s1 = wmma_f16(aq[k], load_b16(kt, CH, 16, k * 32), s1);
        }

        // online softmax update (per D-layout row r; reduce across 16 lanes)
        float scale[8];
#pragma unroll
        for (int r = 0; r < 8; r++) {
            float mx = fmaxf(s0[r], s1[r]);
#pragma unroll
            for (int d = 1; d < 16; d <<= 1) mx = fmaxf(mx, __shfl_xor(mx, d, 32));
            float mnew = fmaxf(mrow[r], mx);
            float sc = __expf(mrow[r] - mnew);
            float p0 = __expf(s0[r] - mnew);
            float p1 = __expf(s1[r] - mnew);
            float rs = p0 + p1;
#pragma unroll
            for (int d = 1; d < 16; d <<= 1) rs += __shfl_xor(rs, d, 32);
            lrow[r] = lrow[r] * sc + rs;
            mrow[r] = mnew;
            scale[r] = sc;
            s0[r] = p0; s1[r] = p1;
        }
#pragma unroll
        for (int c = 0; c < 16; c++) {
#pragma unroll
            for (int r = 0; r < 8; r++) o[c][r] *= scale[r];
        }

        // P (D-layout) -> LDS -> A-fragment layout
#pragma unroll
        for (int r = 0; r < 8; r++) {
            pb[(r + 8 * colh) * 32 + col]      = (_Float16)s0[r];
            pb[(r + 8 * colh) * 32 + 16 + col] = (_Float16)s1[r];
        }
        __syncthreads();
        v16h ap = load_a16(pb, 32, 0, 0);

        // O += P x V   (V fragments from LDS, local j contiguous per lane)
#pragma unroll
        for (int c = 0; c < 16; c++)
            o[c] = wmma_f16(ap, load_b16(vt, 32, c * 16, 0), o[c]);

        // next-tile DMA must be complete before buffers swap
        if (wv_id == 0) __builtin_amdgcn_s_wait_tensorcnt(0);
        __syncthreads();
    }

    // normalize and write attention output as (b, n, c) f16
    float inv[8];
#pragma unroll
    for (int r = 0; r < 8; r++) inv[r] = 1.f / lrow[r];
#pragma unroll
    for (int c = 0; c < 16; c++) {
#pragma unroll
        for (int r = 0; r < 8; r++) {
            int row = i0 + r + 8 * colh;
            ht[(size_t)b * PLANE + (size_t)row * CH + c * 16 + col] =
                (_Float16)(o[c][r] * inv[r]);
        }
    }
}

// ---------------- output projection + residual (WMMA, fp32 out) ----------------
// grid (8, 16, 8), block 256; wave computes 16x64.
__global__ void __launch_bounds__(256) proj_out_kernel(
        const _Float16* w16, const _Float16* ht, const float* bp,
        const float* x, const float* gamma, float* out) {
    int wave = threadIdx.x >> 5, l = threadIdx.x & 31;
    int b  = blockIdx.z;
    int o0 = blockIdx.x * 32 + (wave & 1) * 16;
    int n0 = blockIdx.y * 256 + (wave >> 1) * 64;
    const _Float16* A  = w16 + 196608;               // wp
    const _Float16* Bm = ht + (size_t)b * PLANE;

    v8f acc[4] = {};
    for (int k = 0; k < CH; k += 32) {
        v16h a = load_a16(A, CH, o0, k);
#pragma unroll
        for (int t = 0; t < 4; t++)
            acc[t] = wmma_f16(a, load_b16(Bm, CH, n0 + t * 16, k), acc[t]);
    }
    float g = gamma[0];
    int colh = l >> 4, col = l & 15;
#pragma unroll
    for (int t = 0; t < 4; t++) {
#pragma unroll
        for (int r = 0; r < 8; r++) {
            int m = o0 + r + 8 * colh;
            int n = n0 + t * 16 + col;
            size_t idx = (size_t)b * PLANE + (size_t)m * HW + n;
            out[idx] = x[idx] + g * (acc[t][r] + bp[m]);
        }
    }
}

// ---------------- host launch ----------------
extern "C" void kernel_launch(void* const* d_in, const int* in_sizes, int n_in,
                              void* d_out, int out_size, void* d_ws, size_t ws_size,
                              hipStream_t stream) {
    (void)in_sizes; (void)n_in; (void)out_size; (void)ws_size;
    const float* x    = (const float*)d_in[0];
    const float* gn_w = (const float*)d_in[1];
    const float* gn_b = (const float*)d_in[2];
    const float* wq   = (const float*)d_in[3];
    const float* bq   = (const float*)d_in[4];
    const float* wk   = (const float*)d_in[5];
    const float* bk   = (const float*)d_in[6];
    const float* wv   = (const float*)d_in[7];
    const float* bv   = (const float*)d_in[8];
    const float* wp   = (const float*)d_in[9];
    const float* bp   = (const float*)d_in[10];
    const float* gamma= (const float*)d_in[11];
    float* out = (float*)d_out;

    char* ws = (char*)d_ws;
    _Float16* w16   = (_Float16*)(ws + OFF_W16);
    float*    stats = (float*)   (ws + OFF_STATS);
    _Float16* xnt   = (_Float16*)(ws + OFF_XNT);
    _Float16* q16   = (_Float16*)(ws + OFF_Q);
    _Float16* k16   = (_Float16*)(ws + OFF_K);
    _Float16* v16   = (_Float16*)(ws + OFF_V);
    _Float16* ht    = (_Float16*)(ws + OFF_H);

    conv_w_kernel<<<256, 256, 0, stream>>>(wq, wk, wv, wp, w16);
    gn_stats_kernel<<<BATCH * 32, 256, 0, stream>>>(x, stats);
    gn_apply_kernel<<<(BATCH * PLANE) / 256, 256, 0, stream>>>(x, gn_w, gn_b, stats, xnt);
    qkv_gemm_kernel<<<dim3(8, 16, BATCH * 3), 256, 0, stream>>>(
        w16, xnt, bq, bk, bv, q16, k16, v16);
    flash_attn_kernel<<<dim3(HW / 64, BATCH), 128, 0, stream>>>(q16, k16, v16, ht);
    proj_out_kernel<<<dim3(8, 16, BATCH), 256, 0, stream>>>(w16, ht, bp, x, gamma, out);
}
